// SelfAttention_67637144977482
// MI455X (gfx1250) — matrix operand order repaired and checked by reading, not compiled, not run on hardware
//
#include <hip/hip_runtime.h>

// ---------------------------------------------------------------------------
// MI455X (gfx1250) channel-attention pipeline.
//
// Roofline: ~86 GFLOP, ~270 MB compulsory HBM traffic -> compute-ish bound at
// dense-bf16 WMMA rates, so every GEMM uses v_wmma_f32_16x16x32_bf16 (f32
// accumulate).  All GEMMs are phrased NT (A[M,K] row-major, B[N,K] row-major)
// so BOTH WMMA operands are per-lane K-contiguous loads, using k = 16*half + s
// in both A(16x32) and B(32x16), not the CDNA5 table order.
//
// Main GEMM loop is 2-stage pipelined: global loads for K-tile s+1 are issued
// into staging registers before the DS-read/WMMA phase of tile s, and the
// s_wait_loadcnt is deferred to the LDS store at iteration end (double-
// buffered LDS, 2 x 20KB of the 320KB WGP LDS).
//
// Reference reshape trick: since 4096 = 16*256, the transpose(0,2,1).reshape
// scramble collapses to  scrambled[b,o,m] = out[b, m%256, o*16 + m/256],
// handled as a gather on the B operand of the last GEMM (O fits in the 192MB
// global L2, so the strided re-reads are cache hits).
// ---------------------------------------------------------------------------

typedef __attribute__((ext_vector_type(16))) __bf16 v16bf;
typedef __attribute__((ext_vector_type(8)))  __bf16 v8bf;
typedef __attribute__((ext_vector_type(8)))  float  v8f;

#define NB    16      // batch
#define CIN   512
#define COUT  256
#define NPIX  4096    // 64*64

__device__ __forceinline__ __bf16 f2bf(float f) {
  unsigned u = __builtin_bit_cast(unsigned, f);
  u += 0x7FFFu + ((u >> 16) & 1u);          // round-to-nearest-even
  unsigned short h = (unsigned short)(u >> 16);
  return __builtin_bit_cast(__bf16, h);
}

__device__ __forceinline__ void store_val(float* p, float v)  { *p = v; }
__device__ __forceinline__ void store_val(__bf16* p, float v) { *p = f2bf(v); }

// ---------------------------------------------------------------------------
// f32 -> bf16 elementwise (weights)
// ---------------------------------------------------------------------------
__global__ void cvt_f32_bf16(const float* __restrict__ s, __bf16* __restrict__ d, int n) {
  int i = blockIdx.x * 256 + threadIdx.x;
  if (i < n) d[i] = f2bf(s[i]);
}

// ---------------------------------------------------------------------------
// x[b, c, n] f32  ->  xT[b, n, c] bf16   (LDS-tiled 64x64 transpose)
// ---------------------------------------------------------------------------
__global__ __launch_bounds__(256) void transpose_cvt(const float* __restrict__ x,
                                                     __bf16* __restrict__ xt) {
  __shared__ float tile[64][65];
  const int b = blockIdx.z, n0 = blockIdx.x * 64, c0 = blockIdx.y * 64;
  const float* xp = x + (size_t)b * CIN * NPIX;
  const int t = threadIdx.x;
#pragma unroll
  for (int i = 0; i < 16; ++i) {
    int idx = t + i * 256;
    int c = idx >> 6, n = idx & 63;
    tile[c][n] = xp[(size_t)(c0 + c) * NPIX + n0 + n];
  }
  __syncthreads();
  const int nl = t >> 2, cb = (t & 3) * 16;
  __bf16* dst = xt + (size_t)b * NPIX * CIN + (size_t)(n0 + nl) * CIN + c0 + cb;
  v8bf o0, o1;
#pragma unroll
  for (int j = 0; j < 8; ++j) {
    o0[j] = f2bf(tile[cb + j][nl]);
    o1[j] = f2bf(tile[cb + 8 + j][nl]);
  }
  *(v8bf*)dst       = o0;
  *(v8bf*)(dst + 8) = o1;
}

// ---------------------------------------------------------------------------
// Generic NT bf16 GEMM:  C[M,N] = alpha * A[M,K] * B[N,K]^T (+ bias[m])
//   block tile 128x128, 8 waves (4x2), wave tile 32x64 (2x4 wmma accums)
//   2-stage software pipeline, double-buffered LDS
//   TRANS    : store C transposed (C[n*ldc + m]) -- used to build V^T
//   SCRAMBLE : B[n][k] = Bm[(n&255)*4096 + k*16 + (n>>8)]  (final projection)
// ---------------------------------------------------------------------------
template <typename OutT, bool TRANS, bool SCRAMBLE>
__global__ __launch_bounds__(256) void gemm_nt(
    const __bf16* __restrict__ A,  long long sA, int lda,
    const __bf16* __restrict__ Bm, long long sB, int ldb,
    OutT* __restrict__ C,          long long sC, int ldc,
    const float* __restrict__ bias, float alpha, int K) {
  __shared__ __bf16 As[2][128][40];   // +8 bf16 pad: 80B row stride, 16B aligned
  __shared__ __bf16 Bs[2][128][40];

  const int n0 = blockIdx.x * 128, m0 = blockIdx.y * 128, b = blockIdx.z;
  const __bf16* Ag = A + (size_t)b * sA + (size_t)m0 * lda;
  const __bf16* Bg = Bm + (size_t)b * sB + (SCRAMBLE ? (size_t)0 : (size_t)n0 * ldb);

  const int t = threadIdx.x;
  const int wave = t >> 5, lane = t & 31;
  const int wr = wave >> 1;          // 0..3  (M direction)
  const int wc = wave & 1;           // 0..1  (N direction)
  const int half = lane >> 4, lm = lane & 15;

  // cooperative-load task geometry: 512 16B chunks over 256 threads (x2)
  int arow[2], ack[2];
#pragma unroll
  for (int i = 0; i < 2; ++i) {
    int task = t + i * 256;
    arow[i] = task >> 2;
    ack[i]  = (task & 3) * 8;
  }
  // scramble-gather geometry (one 16-elem K chunk per thread)
  const int srow = t >> 1, skb = (t & 1) * 16;
  const int nn = n0 + srow;
  const __bf16* Sg = Bg + (size_t)(nn & 255) * 4096 + (nn >> 8);

  v8bf areg[2], breg[2];
  __bf16 sreg[16];

  auto ld_glob = [&](int k0) {
#pragma unroll
    for (int i = 0; i < 2; ++i)
      areg[i] = *(const v8bf*)(Ag + (size_t)arow[i] * lda + k0 + ack[i]);
    if constexpr (!SCRAMBLE) {
#pragma unroll
      for (int i = 0; i < 2; ++i)
        breg[i] = *(const v8bf*)(Bg + (size_t)arow[i] * ldb + k0 + ack[i]);
    } else {
#pragma unroll
      for (int j = 0; j < 16; ++j)
        sreg[j] = Sg[(size_t)(k0 + skb + j) * 16];
    }
  };
  auto st_lds = [&](int buf) {
#pragma unroll
    for (int i = 0; i < 2; ++i)
      *(v8bf*)&As[buf][arow[i]][ack[i]] = areg[i];
    if constexpr (!SCRAMBLE) {
#pragma unroll
      for (int i = 0; i < 2; ++i)
        *(v8bf*)&Bs[buf][arow[i]][ack[i]] = breg[i];
    } else {
#pragma unroll
      for (int j = 0; j < 16; ++j)
        Bs[buf][srow][skb + j] = sreg[j];
    }
  };

  v8f acc[2][4] = {};
  const int nk = K >> 5;

  ld_glob(0);
  st_lds(0);

  for (int s = 0; s < nk; ++s) {
    __syncthreads();                     // buffer (s&1) now visible to all
    const int cur = s & 1;
    if (s + 1 < nk) ld_glob((s + 1) << 5);   // prefetch next K-tile (in flight)

    // ---- build WMMA fragments (same contiguous k order in A and B) ----
    // A 16x32: lane holds row m=lm; contiguous K chunk of 16 at 16*half
    v16bf af[2];
#pragma unroll
    for (int mi = 0; mi < 2; ++mi) {
      int row = wr * 32 + mi * 16 + lm;
      af[mi] = *(const v16bf*)&As[cur][row][16 * half];
    }
    // B 32x16: lane holds column n=lm; contiguous K chunk of 16 at 16*half
    v16bf bfr[4];
#pragma unroll
    for (int ni = 0; ni < 4; ++ni) {
      int col = wc * 64 + ni * 16 + lm;
      bfr[ni] = *(const v16bf*)&Bs[cur][col][16 * half];
    }
#pragma unroll
    for (int mi = 0; mi < 2; ++mi)
#pragma unroll
      for (int ni = 0; ni < 4; ++ni)
        acc[mi][ni] = __builtin_amdgcn_wmma_f32_16x16x32_bf16(
            false, af[mi], false, bfr[ni], (short)0, acc[mi][ni], false, false);

    if (s + 1 < nk) st_lds((s + 1) & 1);     // waits loadcnt here, not earlier
  }

  // ---- epilogue: D layout lane=(n=lm), rows m = 8*half + r ----
  OutT* Cg = C + (size_t)b * sC;
#pragma unroll
  for (int mi = 0; mi < 2; ++mi) {
    int gmb = m0 + wr * 32 + mi * 16 + 8 * half;
#pragma unroll
    for (int ni = 0; ni < 4; ++ni) {
      int gn = n0 + wc * 64 + ni * 16 + lm;
      if constexpr (TRANS) {
        v8bf o;
#pragma unroll
        for (int r = 0; r < 8; ++r) {
          float v = acc[mi][ni][r] * alpha;
          if (bias) v += bias[gmb + r];
          o[r] = f2bf(v);
        }
        *(v8bf*)((__bf16*)Cg + (size_t)gn * ldc + gmb) = o;  // 16B store
      } else {
#pragma unroll
        for (int r = 0; r < 8; ++r) {
          float v = acc[mi][ni][r] * alpha;
          if (bias) v += bias[gmb + r];
          store_val(Cg + (size_t)(gmb + r) * ldc + gn, v);
        }
      }
    }
  }
}

// ---------------------------------------------------------------------------
// row softmax: scores f32 [4096 rows x 256]  ->  attn bf16
// ---------------------------------------------------------------------------
__global__ __launch_bounds__(256) void softmax_rows(const float* __restrict__ S,
                                                    __bf16* __restrict__ A) {
  __shared__ float red[256];
  const int row = blockIdx.x, t = threadIdx.x;
  float v = S[(size_t)row * 256 + t];
  red[t] = v;
  __syncthreads();
  for (int s = 128; s > 0; s >>= 1) {
    if (t < s) red[t] = fmaxf(red[t], red[t + s]);
    __syncthreads();
  }
  float mx = red[0];
  __syncthreads();
  float e = __expf(v - mx);
  red[t] = e;
  __syncthreads();
  for (int s = 128; s > 0; s >>= 1) {
    if (t < s) red[t] += red[t + s];
    __syncthreads();
  }
  A[(size_t)row * 256 + t] = f2bf(e / red[0]);
}

// ---------------------------------------------------------------------------
// launch
// ---------------------------------------------------------------------------
extern "C" void kernel_launch(void* const* d_in, const int* in_sizes, int n_in,
                              void* d_out, int out_size, void* d_ws, size_t ws_size,
                              hipStream_t stream) {
  (void)in_sizes; (void)n_in; (void)out_size; (void)ws_size;
  const float* x  = (const float*)d_in[0];
  const float* Wq = (const float*)d_in[1];
  const float* bq = (const float*)d_in[2];
  const float* Wk = (const float*)d_in[3];
  const float* bk = (const float*)d_in[4];
  const float* Wv = (const float*)d_in[5];
  const float* bv = (const float*)d_in[6];
  const float* Wo = (const float*)d_in[7];
  const float* bo = (const float*)d_in[8];

  char* ws = (char*)d_ws;
  size_t off = 0;
  __bf16* Wqb = (__bf16*)(ws + off); off += (size_t)COUT * CIN * 2;       // 256K
  __bf16* Wkb = (__bf16*)(ws + off); off += (size_t)COUT * CIN * 2;
  __bf16* Wvb = (__bf16*)(ws + off); off += (size_t)COUT * CIN * 2;
  __bf16* Wob = (__bf16*)(ws + off); off += (size_t)CIN * COUT * 2;
  __bf16* XT  = (__bf16*)(ws + off); off += (size_t)NB * NPIX * CIN * 2;  // 64MB
  __bf16* Qb  = (__bf16*)(ws + off); off += (size_t)NB * COUT * NPIX * 2; // 32MB
  __bf16* Kb  = (__bf16*)(ws + off); off += (size_t)NB * COUT * NPIX * 2;
  __bf16* Vt  = (__bf16*)(ws + off); off += (size_t)NB * NPIX * COUT * 2; // V^T [N,Cout]
  float*  Sf  = (float*)(ws + off);  off += (size_t)NB * COUT * COUT * 4;
  __bf16* Ab  = (__bf16*)(ws + off); off += (size_t)NB * COUT * COUT * 2;
  __bf16* Ob  = (__bf16*)(ws + off); off += (size_t)NB * COUT * NPIX * 2;

  // 1) weights -> bf16
  cvt_f32_bf16<<<512, 256, 0, stream>>>(Wq, Wqb, COUT * CIN);
  cvt_f32_bf16<<<512, 256, 0, stream>>>(Wk, Wkb, COUT * CIN);
  cvt_f32_bf16<<<512, 256, 0, stream>>>(Wv, Wvb, COUT * CIN);
  cvt_f32_bf16<<<512, 256, 0, stream>>>(Wo, Wob, CIN * COUT);

  // 2) x -> x^T bf16  [B, N, Cin]
  transpose_cvt<<<dim3(NPIX / 64, CIN / 64, NB), 256, 0, stream>>>(x, XT);

  const long long sXT = (long long)NPIX * CIN;
  const long long sQ  = (long long)COUT * NPIX;
  const long long sVT = (long long)NPIX * COUT;

  // 3) Q = Wq * x  [Cout, N] ; K likewise ; V stored transposed -> V^T [N, Cout]
  gemm_nt<__bf16, false, false><<<dim3(32, 2, NB), 256, 0, stream>>>(
      Wqb, 0, CIN, XT, sXT, CIN, Qb, sQ, NPIX, bq, 1.0f, CIN);
  gemm_nt<__bf16, false, false><<<dim3(32, 2, NB), 256, 0, stream>>>(
      Wkb, 0, CIN, XT, sXT, CIN, Kb, sQ, NPIX, bk, 1.0f, CIN);
  gemm_nt<__bf16, true,  false><<<dim3(32, 2, NB), 256, 0, stream>>>(
      Wvb, 0, CIN, XT, sXT, CIN, Vt, sVT, COUT, bv, 1.0f, CIN);

  // 4) scores = (Q K^T) / sqrt(N)   [Cout, Cout] f32
  gemm_nt<float, false, false><<<dim3(2, 2, NB), 256, 0, stream>>>(
      Qb, sQ, NPIX, Kb, sQ, NPIX, Sf, (long long)COUT * COUT, COUT,
      nullptr, 1.0f / 64.0f, NPIX);

  // 5) softmax over key-channel dim
  softmax_rows<<<NB * COUT, 256, 0, stream>>>(Sf, Ab);

  // 6) O = attn * V   [Cout, N] bf16  (B operand = V^T, NT form)
  gemm_nt<__bf16, false, false><<<dim3(32, 2, NB), 256, 0, stream>>>(
      Ab, (long long)COUT * COUT, COUT, Vt, sVT, COUT, Ob, sQ, NPIX,
      nullptr, 1.0f, COUT);

  // 7) out = Wo * scrambled(O) + bo   [Cin, N] f32, scramble fused into B gather
  gemm_nt<float, false, true><<<dim3(32, 4, NB), 256, 0, stream>>>(
      Wob, 0, COUT, Ob, sQ, 0, (float*)d_out, (long long)CIN * NPIX, NPIX,
      bo, 1.0f, COUT);
}
